// DKVMN_4114578670069
// MI455X (gfx1250) — matrix operand order repaired and checked by reading, not compiled, not run on hardware
//
#include <hip/hip_runtime.h>
#include <math.h>

// Problem constants (from reference)
#define BB 128
#define TT 200
#define NS 50      // N_SLOTS
#define DK 128     // MK_DIM
#define DV 200     // MV_DIM
#define FD 50      // F_DIM
#define BT (BB*TT) // 25600 rows, divisible by 16
#define KP 224     // DV padded to multiple of 32
#define NP 208     // DV padded to multiple of 16

typedef __attribute__((ext_vector_type(16))) _Float16 v16h;
typedef __attribute__((ext_vector_type(8)))  _Float16 v8h;
typedef __attribute__((ext_vector_type(8)))  float    v8f;

// ---- fragment loaders (ISA 7.12.2 wave32 layouts) ----
// A (16xK tile, f16): lane L -> M = L&15, kh = L>>4.
//   halves needed: [kbase + kh*8 .. +7] and [kbase + 16 + kh*8 .. +7]  (two b128s)
__device__ __forceinline__ v16h load_A_frag(const _Float16* rowp, int kbase, int kh) {
    v8h lo = *(const v8h*)(rowp + kbase + kh * 8);
    v8h hi = *(const v8h*)(rowp + kbase + 16 + kh * 8);
    return __builtin_shufflevector(lo, hi, 0,1,2,3,4,5,6,7,8,9,10,11,12,13,14,15);
}
// B (Kx16 tile, f16, stored n-major): lane L -> N = L&15, kh = L>>4.
//   halves needed: [kbase + kh*8 .. +7] and [kbase + 16 + kh*8 .. +7]  (two b128s)
__device__ __forceinline__ v16h load_B_frag(const _Float16* colp, int kbase, int kh) {
    const _Float16* p = colp + kbase + kh * 8;
    v8h lo = *(const v8h*)p;
    v8h hi = *(const v8h*)(p + 16);
    return __builtin_shufflevector(lo, hi, 0,1,2,3,4,5,6,7,8,9,10,11,12,13,14,15);
}

__device__ __forceinline__ float sigf(float x) { return 1.f / (1.f + __expf(-x)); }

// =====================================================================
// Kernel 0: one-time f32 -> f16 conversions into WMMA-friendly layouts
// =====================================================================
__global__ void __launch_bounds__(256)
k0_prep(const float* __restrict__ Ek, const float* __restrict__ Ev,
        const float* __restrict__ mk, const float* __restrict__ Wf,
        const float* __restrict__ WE, const float* __restrict__ WA,
        _Float16* __restrict__ Ek16,  _Float16* __restrict__ Ev16,
        _Float16* __restrict__ mkT16, _Float16* __restrict__ WfK16,
        _Float16* __restrict__ WfR16, _Float16* __restrict__ WE16,
        _Float16* __restrict__ WA16,  _Float16* __restrict__ r16) {
    const size_t tid = (size_t)blockIdx.x * blockDim.x + threadIdx.x;
    const size_t stride = (size_t)gridDim.x * blockDim.x;

    // Ek: (2001, 128) row-major, direct
    for (size_t i = tid; i < (size_t)2001 * DK; i += stride)
        Ek16[i] = (_Float16)Ek[i];
    // Ev: (4001, 200) -> (4001, 224) zero-padded
    for (size_t i = tid; i < (size_t)4001 * KP; i += stride) {
        size_t r = i / KP, c = i % KP;
        Ev16[i] = (c < DV) ? (_Float16)Ev[r * DV + c] : (_Float16)0.f;
    }
    // mk^T tiles: (64, 128) n-major (mk already (n,k) row-major), pad n
    for (size_t i = tid; i < (size_t)64 * DK; i += stride) {
        size_t n = i / DK, k = i % DK;
        mkT16[i] = (n < NS) ? (_Float16)mk[n * DK + k] : (_Float16)0.f;
    }
    // Wf[200:328,:]^T: (64, 128) n-major
    for (size_t i = tid; i < (size_t)64 * DK; i += stride) {
        size_t n = i / DK, k = i % DK;
        WfK16[i] = (n < FD) ? (_Float16)Wf[(DV + k) * FD + n] : (_Float16)0.f;
    }
    // Wf[:200,:]^T: (64, 224) n-major, pad n and k
    for (size_t i = tid; i < (size_t)64 * KP; i += stride) {
        size_t n = i / KP, k = i % KP;
        WfR16[i] = (n < FD && k < DV) ? (_Float16)Wf[k * FD + n] : (_Float16)0.f;
    }
    // WE^T / WA^T: (208, 224) n-major, pad n and k
    for (size_t i = tid; i < (size_t)NP * KP; i += stride) {
        size_t n = i / KP, k = i % KP;
        bool ok = (n < DV) && (k < DV);
        WE16[i] = ok ? (_Float16)WE[k * DV + n] : (_Float16)0.f;
        WA16[i] = ok ? (_Float16)WA[k * DV + n] : (_Float16)0.f;
    }
    // r16: zero the pad columns (200..223); scan fills the rest
    for (size_t i = tid; i < (size_t)BT * (KP - DV); i += stride) {
        size_t r = i / (KP - DV), c = i % (KP - DV);
        r16[r * KP + DV + c] = (_Float16)0.f;
    }
}

// =====================================================================
// Kernel 1: k = Ek[skill]; w = softmax(k @ mk^T); fk = k @ Wf[200:,:]
// blockDim = 32 (one wave)
// =====================================================================
__global__ void __launch_bounds__(32)
k1_w_fk(const int* __restrict__ inputs, const _Float16* __restrict__ Ek16,
        const _Float16* __restrict__ mkT16, const _Float16* __restrict__ WfK16,
        float* __restrict__ w_all, float* __restrict__ fk_all) {
    __shared__ int   sskill[16];
    __shared__ float sm[16 * 64];
    __shared__ float smax[16], sinv[16];

    const int lane = threadIdx.x;
    const int row0 = blockIdx.x * 16;
    if (lane < 16) sskill[lane] = inputs[(row0 + lane) * 2 + 0];
    __syncthreads();

    const int m  = lane & 15;
    const int kh = lane >> 4;
    const int mh = 8 * kh;

    // A fragments: gathered Ek row for M = m, reused across all N tiles/GEMMs
    const _Float16* arow = Ek16 + (size_t)sskill[m] * DK;
    v16h A[4];
#pragma unroll
    for (int kt = 0; kt < 4; ++kt) A[kt] = load_A_frag(arow, kt * 32, kh);

    // ---- logits = k @ mk^T ----
#pragma unroll
    for (int nt = 0; nt < 4; ++nt) {
        int n = nt * 16 + m;
        const _Float16* bcol = mkT16 + (size_t)n * DK;
        v16h B[4];
#pragma unroll
        for (int kt = 0; kt < 4; ++kt) B[kt] = load_B_frag(bcol, kt * 32, kh);
        v8f acc = {};
#pragma unroll
        for (int kt = 0; kt < 4; ++kt)
            acc = __builtin_amdgcn_wmma_f32_16x16x32_f16(false, A[kt], false, B[kt],
                                                         (short)0, acc, false, false);
#pragma unroll
        for (int i = 0; i < 8; ++i) sm[(i + mh) * 64 + n] = acc[i];
    }
    __syncthreads();

    // ---- row softmax over n<50 ----
    if (lane < 16) {
        float mx = -1e30f;
        for (int n = 0; n < NS; ++n) mx = fmaxf(mx, sm[lane * 64 + n]);
        float s = 0.f;
        for (int n = 0; n < NS; ++n) s += __expf(sm[lane * 64 + n] - mx);
        smax[lane] = mx;
        sinv[lane] = 1.f / s;
    }
    __syncthreads();
    for (int idx = lane; idx < 16 * NS; idx += 32) {
        int r = idx / NS, n = idx % NS;
        w_all[(row0 + r) * NS + n] = __expf(sm[r * 64 + n] - smax[r]) * sinv[r];
    }

    // ---- fk = k @ Wf[200:,:] ----
#pragma unroll
    for (int nt = 0; nt < 4; ++nt) {
        int n = nt * 16 + m;
        const _Float16* bcol = WfK16 + (size_t)n * DK;
        v16h B[4];
#pragma unroll
        for (int kt = 0; kt < 4; ++kt) B[kt] = load_B_frag(bcol, kt * 32, kh);
        v8f acc = {};
#pragma unroll
        for (int kt = 0; kt < 4; ++kt)
            acc = __builtin_amdgcn_wmma_f32_16x16x32_f16(false, A[kt], false, B[kt],
                                                         (short)0, acc, false, false);
        if (n < FD) {
#pragma unroll
            for (int i = 0; i < 8; ++i)
                fk_all[(row0 + i + mh) * FD + n] = acc[i];
        }
    }
}

// =====================================================================
// Kernel 2: v = Ev[swc]; e = sigmoid(v@WE+bE); a = tanh(v@WA+bA)
// blockDim = 64 (2 waves split the 13 N-tiles)
// =====================================================================
__global__ void __launch_bounds__(64)
k2_ea(const int* __restrict__ inputs, const _Float16* __restrict__ Ev16,
      const _Float16* __restrict__ WE16, const _Float16* __restrict__ WA16,
      const float* __restrict__ bE, const float* __restrict__ bA,
      float* __restrict__ e_all, float* __restrict__ a_all) {
    __shared__ int sswc[16];
    const int tid  = threadIdx.x;
    const int lane = tid & 31;
    const int wv   = tid >> 5;
    const int row0 = blockIdx.x * 16;
    if (tid < 16) sswc[tid] = inputs[(row0 + tid) * 2 + 1];
    __syncthreads();

    const int m  = lane & 15;
    const int kh = lane >> 4;
    const int mh = 8 * kh;

    const _Float16* arow = Ev16 + (size_t)sswc[m] * KP;
    v16h A[7];
#pragma unroll
    for (int kt = 0; kt < 7; ++kt) A[kt] = load_A_frag(arow, kt * 32, kh);

    for (int nt = wv; nt < 13; nt += 2) {   // wave-uniform loop, EXEC all-ones
        int n = nt * 16 + m;
        const _Float16* be = WE16 + (size_t)n * KP;
        const _Float16* ba = WA16 + (size_t)n * KP;
        // batch-preload all B fragments so the WMMA chain isn't load-use bound
        v16h BE[7], BA[7];
#pragma unroll
        for (int kt = 0; kt < 7; ++kt) BE[kt] = load_B_frag(be, kt * 32, kh);
#pragma unroll
        for (int kt = 0; kt < 7; ++kt) BA[kt] = load_B_frag(ba, kt * 32, kh);
        v8f accE = {}, accA = {};
#pragma unroll
        for (int kt = 0; kt < 7; ++kt)
            accE = __builtin_amdgcn_wmma_f32_16x16x32_f16(false, A[kt], false, BE[kt],
                                                          (short)0, accE, false, false);
#pragma unroll
        for (int kt = 0; kt < 7; ++kt)
            accA = __builtin_amdgcn_wmma_f32_16x16x32_f16(false, A[kt], false, BA[kt],
                                                          (short)0, accA, false, false);
        if (n < DV) {
            float vbE = bE[n], vbA = bA[n];
#pragma unroll
            for (int i = 0; i < 8; ++i) {
                int row = row0 + i + mh;
                e_all[row * DV + n] = sigf(accE[i] + vbE);
                a_all[row * DV + n] = tanhf(accA[i] + vbA);
            }
        }
    }
}

// =====================================================================
// Kernel 3: sequential scan. One block per batch element; mv[50] per thread
// in registers (thread d owns column d). Writes r directly as f16 (padded).
// =====================================================================
__global__ void __launch_bounds__(256)
k3_scan(const float* __restrict__ w_all, const float* __restrict__ e_all,
        const float* __restrict__ a_all, _Float16* __restrict__ r16) {
    __shared__ float wsh[NS];
    const int b = blockIdx.x;
    const int d = threadIdx.x;
    const bool act = d < DV;

    float mv[NS];
#pragma unroll
    for (int n = 0; n < NS; ++n) mv[n] = 0.f;

    for (int t = 0; t < TT; ++t) {
        const int row = b * TT + t;
        if (threadIdx.x < NS) wsh[threadIdx.x] = w_all[row * NS + threadIdx.x];
        __syncthreads();
        if (act) {
            float e = e_all[row * DV + d];
            float a = a_all[row * DV + d];
            float r = 0.f;
#pragma unroll
            for (int n = 0; n < NS; ++n) {
                float wn = wsh[n];
                r = fmaf(wn, mv[n], r);                     // read state BEFORE update
                mv[n] = fmaf(wn, a, fmaf(-wn * e, mv[n], mv[n]));
            }
            r16[(size_t)row * KP + d] = (_Float16)r;
        }
        __syncthreads();
    }
}

// =====================================================================
// Kernel 4: f = tanh(r @ Wf[:200,:] + fk + bf); p = sigmoid(f @ Wp + bp)
// blockDim = 32 (one wave)
// =====================================================================
__global__ void __launch_bounds__(32)
k4_pred(const _Float16* __restrict__ r16, const float* __restrict__ fk_all,
        const _Float16* __restrict__ WfR16, const float* __restrict__ bf,
        const float* __restrict__ Wp, const float* __restrict__ bp,
        float* __restrict__ out) {
    __shared__ float fsh[16 * 64];
    const int lane = threadIdx.x;
    const int row0 = blockIdx.x * 16;
    const int m  = lane & 15;
    const int kh = lane >> 4;
    const int mh = 8 * kh;

    const _Float16* arow = r16 + (size_t)(row0 + m) * KP;
    v16h A[7];
#pragma unroll
    for (int kt = 0; kt < 7; ++kt) A[kt] = load_A_frag(arow, kt * 32, kh);

#pragma unroll
    for (int nt = 0; nt < 4; ++nt) {
        int n = nt * 16 + m;
        const _Float16* bcol = WfR16 + (size_t)n * KP;
        v16h B[7];
#pragma unroll
        for (int kt = 0; kt < 7; ++kt) B[kt] = load_B_frag(bcol, kt * 32, kh);
        v8f acc = {};
#pragma unroll
        for (int kt = 0; kt < 7; ++kt)
            acc = __builtin_amdgcn_wmma_f32_16x16x32_f16(false, A[kt], false, B[kt],
                                                         (short)0, acc, false, false);
        float vbf = (n < FD) ? bf[n] : 0.f;
#pragma unroll
        for (int i = 0; i < 8; ++i) {
            int mm = i + mh;
            float fval = 0.f;
            if (n < FD) fval = tanhf(acc[i] + fk_all[(row0 + mm) * FD + n] + vbf);
            fsh[mm * 64 + n] = fval;
        }
    }
    __syncthreads();

    if (lane < 16) {
        float acc = bp[0];
        for (int n = 0; n < FD; ++n) acc = fmaf(fsh[lane * 64 + n], Wp[n], acc);
        out[row0 + lane] = sigf(acc);
    }
}

// =====================================================================
extern "C" void kernel_launch(void* const* d_in, const int* in_sizes, int n_in,
                              void* d_out, int out_size, void* d_ws, size_t ws_size,
                              hipStream_t stream) {
    const int*   inputs = (const int*)  d_in[0];
    const float* Ek     = (const float*)d_in[1];
    const float* Ev     = (const float*)d_in[2];
    const float* mk     = (const float*)d_in[3];
    const float* Wf     = (const float*)d_in[4];
    const float* bf     = (const float*)d_in[5];
    const float* Wp     = (const float*)d_in[6];
    const float* bp     = (const float*)d_in[7];
    const float* WE     = (const float*)d_in[8];
    const float* bE     = (const float*)d_in[9];
    const float* WA     = (const float*)d_in[10];
    const float* bA     = (const float*)d_in[11];
    float* out = (float*)d_out;

    // ---- workspace layout ----
    char* ws = (char*)d_ws;
    size_t off = 0;
    auto alloc = [&](size_t bytes) {
        char* p = ws + off;
        off += (bytes + 255) & ~(size_t)255;   // keep 256B alignment
        return p;
    };
    float* w_all  = (float*)alloc((size_t)BT * NS * 4);
    float* fk_all = (float*)alloc((size_t)BT * FD * 4);
    float* e_all  = (float*)alloc((size_t)BT * DV * 4);
    float* a_all  = (float*)alloc((size_t)BT * DV * 4);
    _Float16* Ek16  = (_Float16*)alloc((size_t)2001 * DK * 2);
    _Float16* Ev16  = (_Float16*)alloc((size_t)4001 * KP * 2);
    _Float16* mkT16 = (_Float16*)alloc((size_t)64 * DK * 2);
    _Float16* WfK16 = (_Float16*)alloc((size_t)64 * DK * 2);
    _Float16* WfR16 = (_Float16*)alloc((size_t)64 * KP * 2);
    _Float16* WE16  = (_Float16*)alloc((size_t)NP * KP * 2);
    _Float16* WA16  = (_Float16*)alloc((size_t)NP * KP * 2);
    _Float16* r16   = (_Float16*)alloc((size_t)BT * KP * 2);

    const int tiles = BT / 16;  // 1600

    k0_prep<<<1024, 256, 0, stream>>>(Ek, Ev, mk, Wf, WE, WA,
                                      Ek16, Ev16, mkT16, WfK16, WfR16, WE16, WA16, r16);
    k1_w_fk<<<tiles, 32, 0, stream>>>(inputs, Ek16, mkT16, WfK16, w_all, fk_all);
    k2_ea  <<<tiles, 64, 0, stream>>>(inputs, Ev16, WE16, WA16, bE, bA, e_all, a_all);
    k3_scan<<<BB, 256, 0, stream>>>(w_all, e_all, a_all, r16);
    k4_pred<<<tiles, 32, 0, stream>>>(r16, fk_all, WfR16, bf, Wp, bp, out);
}
